// cylinder_fea_11974368821427
// MI455X (gfx1250) — matrix operand order repaired and checked by reading, not compiled, not run on hardware
//
#include <hip/hip_runtime.h>

// ---------------------------------------------------------------------------
// Types / helpers
// ---------------------------------------------------------------------------
typedef unsigned short u16_t;
typedef __bf16 bf16_t;
typedef bf16_t v16bf __attribute__((ext_vector_type(16)));
typedef float  f32x8 __attribute__((ext_vector_type(8)));
typedef unsigned int u32x4 __attribute__((ext_vector_type(4)));

__device__ __forceinline__ u16_t f2bf(float f) {
  unsigned u = __float_as_uint(f);
  unsigned r = u + 0x7FFFu + ((u >> 16) & 1u);   // round-to-nearest-even
  return (u16_t)(r >> 16);
}

union AFrag { v16bf bf; u32x4 q[2]; };

__device__ __forceinline__ f32x8 wmma_bf16(const AFrag& a, const AFrag& b, f32x8 c) {
  return __builtin_amdgcn_wmma_f32_16x16x32_bf16(false, a.bf, false, b.bf,
                                                 (short)0, c, false, false);
}

// A-matrix 16x32 bf16 fragment from row-major global memory.
// Lane L<16 holds row L, K in {kb..kb+7, kb+16..kb+23}, kb = (L>>4)*8 (+32*kstep)
// -> two contiguous 16B chunks per lane.
__device__ __forceinline__ void load_a_global(AFrag& a, const u16_t* tileBase,
                                              int rowLen, int kstep, int lane) {
  int row = lane & 15;
  int kb  = ((lane >> 4) << 3) + kstep * 32;
  const u16_t* p = tileBase + (size_t)row * rowLen + kb;
  a.q[0] = __builtin_nontemporal_load((const u32x4*)p);
  a.q[1] = __builtin_nontemporal_load((const u32x4*)(p + 16));
}

// Same layout but source tile lives in LDS (16 x 32 bf16, row stride 32).
__device__ __forceinline__ void load_a_lds(AFrag& a, const u16_t* stage, int lane) {
  int row = lane & 15;
  int kb  = ((lane >> 4) << 3);
  const u16_t* p = stage + row * 32 + kb;
  a.q[0] = *(const u32x4*)p;
  a.q[1] = *(const u32x4*)(p + 16);
}

// B-matrix 32x16 bf16 fragment from pre-transposed weights Wt[NOUT][KIN]:
// lane L holds column (L&15), K in {kb..kb+7, kb+16..kb+23}, kb = (L>>4)*8 + 32*kstep
// -> two contiguous 16B chunks per lane.
__device__ __forceinline__ void load_b_global(AFrag& b, const u16_t* Wt,
                                              int kin, int nf, int kstep, int lane) {
  int col = lane & 15;
  int kb  = ((lane >> 4) << 3) + kstep * 32;
  const u16_t* p = Wt + (size_t)(nf * 16 + col) * kin + kb;
  b.q[0] = *(const u32x4*)p;
  b.q[1] = *(const u32x4*)(p + 16);
}

// CDNA5 float atomic max (GLOBAL_ATOMIC_MAX_NUM_F32, non-returning).
__device__ __forceinline__ void atomic_max_f32(float* addr, float v) {
  asm volatile("global_atomic_max_num_f32 %0, %1, off" :: "v"(addr), "v"(v) : "memory");
}

// ---------------------------------------------------------------------------
// K0: init — zero stats region of ws, fill pooled output with -inf
// ---------------------------------------------------------------------------
__global__ __launch_bounds__(256) void init_kernel(float* wsf, float* pooled, int pooledN) {
  int i = blockIdx.x * blockDim.x + threadIdx.x;
  int stride = gridDim.x * blockDim.x;
  for (int idx = i; idx < 2048; idx += stride) wsf[idx] = 0.0f;
  const float ninf = __int_as_float(0xff800000);
  for (int idx = i; idx < pooledN; idx += stride) pooled[idx] = ninf;
}

// ---------------------------------------------------------------------------
// K-prep: transpose + convert f32 weights [kin][nout] -> bf16 Wt[nout][kin]
// ---------------------------------------------------------------------------
__global__ __launch_bounds__(256) void transpose_cvt_kernel(const float* W, u16_t* Wt,
                                                            int kin, int nout) {
  int idx = blockIdx.x * blockDim.x + threadIdx.x;
  if (idx < kin * nout) {
    int k = idx / nout, n = idx % nout;
    Wt[(size_t)n * kin + k] = f2bf(W[idx]);
  }
}

// ---------------------------------------------------------------------------
// K1: per-feature sum / sumsq of input x [N,3]
// ---------------------------------------------------------------------------
__global__ __launch_bounds__(256) void stats0_kernel(const float* X, float* gsum,
                                                     float* gsq, int nrows) {
  __shared__ float ls[3], lq[3];
  if (threadIdx.x < 3) { ls[threadIdx.x] = 0.f; lq[threadIdx.x] = 0.f; }
  __syncthreads();
  float s0 = 0, s1 = 0, s2 = 0, q0 = 0, q1 = 0, q2 = 0;
  for (int r = blockIdx.x * blockDim.x + threadIdx.x; r < nrows;
       r += gridDim.x * blockDim.x) {
    float x0 = X[(size_t)r * 3 + 0], x1 = X[(size_t)r * 3 + 1], x2 = X[(size_t)r * 3 + 2];
    s0 += x0; s1 += x1; s2 += x2;
    q0 += x0 * x0; q1 += x1 * x1; q2 += x2 * x2;
  }
  atomicAdd(&ls[0], s0); atomicAdd(&ls[1], s1); atomicAdd(&ls[2], s2);
  atomicAdd(&lq[0], q0); atomicAdd(&lq[1], q1); atomicAdd(&lq[2], q2);
  __syncthreads();
  if (threadIdx.x < 3) {
    atomicAdd(&gsum[threadIdx.x], ls[threadIdx.x]);
    atomicAdd(&gsq[threadIdx.x],  lq[threadIdx.x]);
  }
}

// ---------------------------------------------------------------------------
// Finalize BN fold: a = g*rsqrt(var+eps); c = beta - mean*a (+ b*a if bias given)
// ---------------------------------------------------------------------------
__global__ void finalize_kernel(const float* sum, const float* sq, const float* g,
                                const float* beta, const float* b,
                                float* a_out, float* c_out, int n, float invN) {
  int t = blockIdx.x * blockDim.x + threadIdx.x;
  if (t < n) {
    float m = sum[t] * invN;
    float v = fmaxf(sq[t] * invN - m * m, 0.0f);
    float a = g[t] * rsqrtf(v + 1e-5f);
    float c = beta[t] - m * a;
    if (b) c += b[t] * a;
    a_out[t] = a; c_out[t] = c;
  }
}

// ---------------------------------------------------------------------------
// K2: layer-1 stats (K=3 -> scalar VALU). y1 = (x*a0+c0)@W1 + b1, reduce sum/sq
// ---------------------------------------------------------------------------
__global__ __launch_bounds__(128) void layer1_stats_kernel(
    const float* X, const float* W1, const float* b1,
    const float* a0, const float* c0, float* gsum, float* gsq, int nrows) {
  __shared__ float sW[192], sb[64], ls[64], lq[64], sa0[3], sc0[3];
  int t = threadIdx.x;
  for (int i = t; i < 192; i += blockDim.x) sW[i] = W1[i];
  for (int i = t; i < 64; i += blockDim.x) { sb[i] = b1[i]; ls[i] = 0.f; lq[i] = 0.f; }
  if (t < 3) { sa0[t] = a0[t]; sc0[t] = c0[t]; }
  __syncthreads();
  float s[64], q[64];
#pragma unroll
  for (int j = 0; j < 64; j++) { s[j] = 0.f; q[j] = 0.f; }
  for (int r = blockIdx.x * blockDim.x + t; r < nrows; r += gridDim.x * blockDim.x) {
    float u0 = X[(size_t)r * 3 + 0] * sa0[0] + sc0[0];
    float u1 = X[(size_t)r * 3 + 1] * sa0[1] + sc0[1];
    float u2 = X[(size_t)r * 3 + 2] * sa0[2] + sc0[2];
#pragma unroll
    for (int j = 0; j < 64; j++) {
      float y = sb[j] + u0 * sW[j] + u1 * sW[64 + j] + u2 * sW[128 + j];
      s[j] += y; q[j] += y * y;
    }
  }
#pragma unroll
  for (int j = 0; j < 64; j++) { atomicAdd(&ls[j], s[j]); atomicAdd(&lq[j], q[j]); }
  __syncthreads();
  for (int i = t; i < 64; i += blockDim.x) {
    atomicAdd(&gsum[i], ls[i]); atomicAdd(&gsq[i], lq[i]);
  }
}

// ---------------------------------------------------------------------------
// K3: layer-1 apply: h1 = relu((x*a0+c0)@W1 * a1 + c1) -> bf16 [N,64]
// ---------------------------------------------------------------------------
__global__ __launch_bounds__(128) void layer1_apply_kernel(
    const float* X, const float* W1, const float* a0, const float* c0,
    const float* a1, const float* c1, u16_t* H1, int nrows) {
  __shared__ float sW[192], sa1[64], sc1[64], sa0[3], sc0[3];
  int t = threadIdx.x;
  for (int i = t; i < 192; i += blockDim.x) sW[i] = W1[i];
  for (int i = t; i < 64; i += blockDim.x) { sa1[i] = a1[i]; sc1[i] = c1[i]; }
  if (t < 3) { sa0[t] = a0[t]; sc0[t] = c0[t]; }
  __syncthreads();
  for (int r = blockIdx.x * blockDim.x + t; r < nrows; r += gridDim.x * blockDim.x) {
    float u0 = X[(size_t)r * 3 + 0] * sa0[0] + sc0[0];
    float u1 = X[(size_t)r * 3 + 1] * sa0[1] + sc0[1];
    float u2 = X[(size_t)r * 3 + 2] * sa0[2] + sc0[2];
    unsigned pk[32];
#pragma unroll
    for (int j = 0; j < 64; j += 2) {
      float ya = u0 * sW[j] + u1 * sW[64 + j] + u2 * sW[128 + j];
      float yb = u0 * sW[j + 1] + u1 * sW[64 + j + 1] + u2 * sW[128 + j + 1];
      float va = fmaxf(ya * sa1[j] + sc1[j], 0.0f);
      float vb = fmaxf(yb * sa1[j + 1] + sc1[j + 1], 0.0f);
      pk[j >> 1] = (unsigned)f2bf(va) | ((unsigned)f2bf(vb) << 16);
    }
    u32x4* outp = (u32x4*)(H1 + (size_t)r * 64);
#pragma unroll
    for (int i = 0; i < 8; i++) {
      u32x4 w = {pk[4 * i], pk[4 * i + 1], pk[4 * i + 2], pk[4 * i + 3]};
      __builtin_nontemporal_store(w, outp + i);
    }
  }
}

// ---------------------------------------------------------------------------
// WMMA stats pass: y = X@W + b, accumulate per-column sum/sumsq (no store)
// ---------------------------------------------------------------------------
template <int KIN, int NOUT>
__global__ __launch_bounds__(128) void gemm_stats_kernel(
    const u16_t* __restrict__ X, const u16_t* __restrict__ Wt,
    const float* __restrict__ bias, float* gsum, float* gsq, int nrows) {
  constexpr int WAVES = 4;
  __shared__ float ls[NOUT], lq[NOUT];
  for (int i = threadIdx.x; i < NOUT; i += blockDim.x) { ls[i] = 0.f; lq[i] = 0.f; }
  __syncthreads();
  int lane = threadIdx.x & 31, wave = threadIdx.x >> 5;
  int ntiles = nrows >> 4;
  int col0 = lane & 15;
  for (int tile = blockIdx.x * WAVES + wave; tile < ntiles; tile += gridDim.x * WAVES) {
    const u16_t* base = X + (size_t)tile * 16 * KIN;
    AFrag a[KIN / 32];
#pragma unroll
    for (int k = 0; k < KIN / 32; k++) load_a_global(a[k], base, KIN, k, lane);
#pragma unroll
    for (int nf = 0; nf < NOUT / 16; nf++) {
      f32x8 acc = {0, 0, 0, 0, 0, 0, 0, 0};
#pragma unroll
      for (int k = 0; k < KIN / 32; k++) {
        AFrag b; load_b_global(b, Wt, KIN, nf, k, lane);
        acc = wmma_bf16(a[k], b, acc);
      }
      float bcol = bias[nf * 16 + col0];
      float s = 0.f, q = 0.f;
#pragma unroll
      for (int r = 0; r < 8; r++) { float v = acc[r] + bcol; s += v; q += v * v; }
      atomicAdd(&ls[nf * 16 + col0], s);
      atomicAdd(&lq[nf * 16 + col0], q);
    }
  }
  __syncthreads();
  for (int i = threadIdx.x; i < NOUT; i += blockDim.x) {
    atomicAdd(&gsum[i], ls[i]); atomicAdd(&gsq[i], lq[i]);
  }
}

// ---------------------------------------------------------------------------
// WMMA apply pass: H = relu((X@W)*a + c) -> bf16 [N,NOUT]
// ---------------------------------------------------------------------------
template <int KIN, int NOUT>
__global__ __launch_bounds__(128) void gemm_apply_kernel(
    const u16_t* __restrict__ X, const u16_t* __restrict__ Wt,
    const float* __restrict__ a_s, const float* __restrict__ c_s,
    u16_t* __restrict__ H, int nrows) {
  constexpr int WAVES = 4;
  int lane = threadIdx.x & 31, wave = threadIdx.x >> 5;
  int ntiles = nrows >> 4;
  int col0 = lane & 15;
  int rbase = (lane >> 4) << 3;
  for (int tile = blockIdx.x * WAVES + wave; tile < ntiles; tile += gridDim.x * WAVES) {
    const u16_t* base = X + (size_t)tile * 16 * KIN;
    AFrag a[KIN / 32];
#pragma unroll
    for (int k = 0; k < KIN / 32; k++) load_a_global(a[k], base, KIN, k, lane);
#pragma unroll
    for (int nf = 0; nf < NOUT / 16; nf++) {
      f32x8 acc = {0, 0, 0, 0, 0, 0, 0, 0};
#pragma unroll
      for (int k = 0; k < KIN / 32; k++) {
        AFrag b; load_b_global(b, Wt, KIN, nf, k, lane);
        acc = wmma_bf16(a[k], b, acc);
      }
      float av = a_s[nf * 16 + col0], cv = c_s[nf * 16 + col0];
#pragma unroll
      for (int r = 0; r < 8; r++) {
        int row = rbase + r;
        float v = fmaxf(acc[r] * av + cv, 0.0f);
        u16_t* p = H + (size_t)(tile * 16 + row) * NOUT + nf * 16 + col0;
        __builtin_nontemporal_store(f2bf(v), p);
      }
    }
  }
}

// ---------------------------------------------------------------------------
// K-final: fused layer3(WMMA,128->256,BN,ReLU) + layer4(WMMA,256->128,+b4)
//          + segment-max via global_atomic_max_num_f32.
// h3 never leaves the WGP: 16x32 chunks restaged through LDS into A-fragments.
// Block-uniform tile loop so barriers and WMMA (EXEC all-1s) are safe.
// ---------------------------------------------------------------------------
__global__ __launch_bounds__(128) void final_fused_kernel(
    const u16_t* __restrict__ H2, const u16_t* __restrict__ W3t,
    const float* __restrict__ a3, const float* __restrict__ c3,
    const u16_t* __restrict__ W4t, const float* __restrict__ b4,
    const int* __restrict__ seg, float* __restrict__ pooled, int nrows) {
  constexpr int WAVES = 4;
  __shared__ __align__(16) u16_t stage[WAVES][16 * 32];
  int lane = threadIdx.x & 31, wave = threadIdx.x >> 5;
  int col0 = lane & 15;
  int rbase = (lane >> 4) << 3;
  int ntiles = nrows >> 4;
  int stride = gridDim.x * WAVES;

  for (int it = 0;; ++it) {
    int first = blockIdx.x * WAVES + it * stride;   // block-uniform
    if (first >= ntiles) break;
    int myTile = first + wave;
    bool active = myTile < ntiles;
    int tl = active ? myTile : (ntiles - 1);

    const u16_t* base = H2 + (size_t)tl * 16 * 128;
    AFrag a2[4];
#pragma unroll
    for (int k = 0; k < 4; k++) load_a_global(a2[k], base, 128, k, lane);

    f32x8 acc4[8];
#pragma unroll
    for (int i = 0; i < 8; i++) acc4[i] = (f32x8){0, 0, 0, 0, 0, 0, 0, 0};

    for (int ch = 0; ch < 8; ch++) {           // 8 chunks of 32 h3-columns
#pragma unroll
      for (int half = 0; half < 2; half++) {   // layer-3: two 16-col fragments
        int nf = ch * 2 + half;
        f32x8 acc = {0, 0, 0, 0, 0, 0, 0, 0};
#pragma unroll
        for (int k = 0; k < 4; k++) {
          AFrag b; load_b_global(b, W3t, 128, nf, k, lane);
          acc = wmma_bf16(a2[k], b, acc);
        }
        float av = a3[nf * 16 + col0], cv = c3[nf * 16 + col0];
        int colL = half * 16 + col0;
#pragma unroll
        for (int r = 0; r < 8; r++) {
          int row = rbase + r;
          float v = fmaxf(acc[r] * av + cv, 0.0f);
          stage[wave][row * 32 + colL] = f2bf(v);
        }
      }
      __syncthreads();
      AFrag a4; load_a_lds(a4, &stage[wave][0], lane);
      __syncthreads();
#pragma unroll
      for (int nf4 = 0; nf4 < 8; nf4++) {      // layer-4: K-chunk ch of 256
        AFrag b; load_b_global(b, W4t, 256, nf4, ch, lane);
        acc4[nf4] = wmma_bf16(a4, b, acc4[nf4]);
      }
    }

    if (active) {
#pragma unroll
      for (int nf4 = 0; nf4 < 8; nf4++) {
        float bb = b4[nf4 * 16 + col0];
#pragma unroll
        for (int r = 0; r < 8; r++) {
          int grow = myTile * 16 + rbase + r;
          int s = seg[grow];
          float v = acc4[nf4][r] + bb;
          atomic_max_f32(pooled + (size_t)s * 128 + nf4 * 16 + col0, v);
        }
      }
    }
  }
}

// ---------------------------------------------------------------------------
// Host launcher
// ---------------------------------------------------------------------------
extern "C" void kernel_launch(void* const* d_in, const int* in_sizes, int n_in,
                              void* d_out, int out_size, void* d_ws, size_t ws_size,
                              hipStream_t stream) {
  const float* pt_fea  = (const float*)d_in[0];
  const int*   unq_inv = (const int*)d_in[3];
  const float* bn0_g = (const float*)d_in[4],  *bn0_b = (const float*)d_in[5];
  const float* bn1_g = (const float*)d_in[6],  *bn1_b = (const float*)d_in[7];
  const float* bn2_g = (const float*)d_in[8],  *bn2_b = (const float*)d_in[9];
  const float* bn3_g = (const float*)d_in[10], *bn3_b = (const float*)d_in[11];
  const float* W1 = (const float*)d_in[12], *b1 = (const float*)d_in[13];
  const float* W2 = (const float*)d_in[14], *b2 = (const float*)d_in[15];
  const float* W3 = (const float*)d_in[16], *b3 = (const float*)d_in[17];
  const float* W4 = (const float*)d_in[18], *b4 = (const float*)d_in[19];

  const int N = in_sizes[0] / 3;
  const int U = in_sizes[2] / 3;
  const float invN = 1.0f / (float)N;

  // ws layout (floats for stats/affine, then bf16 weights, then activations)
  enum { S0 = 0, Q0 = 3, S1 = 6, Q1 = 70, S2 = 134, Q2 = 262, S3 = 390, Q3 = 646,
         A0 = 902, C0 = 905, A1 = 908, C1 = 972, A2 = 1036, C2 = 1164,
         A3 = 1292, C3 = 1548 };
  float* wsf = (float*)d_ws;
  char*  wsb = (char*)d_ws;
  u16_t* W2t = (u16_t*)(wsb + 8192);                      // 128x64  bf16
  u16_t* W3t = (u16_t*)(wsb + 8192 + 16384);              // 256x128 bf16
  u16_t* W4t = (u16_t*)(wsb + 8192 + 16384 + 65536);      // 128x256 bf16
  u16_t* H1  = (u16_t*)(wsb + 163840);                    // N x 64  bf16
  u16_t* H2  = (u16_t*)(wsb + 163840 + (size_t)N * 64 * 2); // N x 128 bf16
  float* pooled = (float*)((char*)d_out + (size_t)U * 24);  // after int64 unq
  const int pooledN = U * 128;

  // unq passthrough (int64 [U,3] raw bytes)
  (void)hipMemcpyAsync(d_out, d_in[2], (size_t)U * 24, hipMemcpyDeviceToDevice, stream);

  // init stats + pooled=-inf
  {
    int total = pooledN > 2048 ? pooledN : 2048;
    init_kernel<<<(total + 255) / 256, 256, 0, stream>>>(wsf, pooled, pooledN);
  }

  // weight transpose+convert
  transpose_cvt_kernel<<<(64 * 128 + 255) / 256, 256, 0, stream>>>(W2, W2t, 64, 128);
  transpose_cvt_kernel<<<(128 * 256 + 255) / 256, 256, 0, stream>>>(W3, W3t, 128, 256);
  transpose_cvt_kernel<<<(256 * 128 + 255) / 256, 256, 0, stream>>>(W4, W4t, 256, 128);

  // bn0
  stats0_kernel<<<1024, 256, 0, stream>>>(pt_fea, wsf + S0, wsf + Q0, N);
  finalize_kernel<<<1, 256, 0, stream>>>(wsf + S0, wsf + Q0, bn0_g, bn0_b, nullptr,
                                         wsf + A0, wsf + C0, 3, invN);
  // layer1 (scalar, K=3)
  layer1_stats_kernel<<<2048, 128, 0, stream>>>(pt_fea, W1, b1, wsf + A0, wsf + C0,
                                                wsf + S1, wsf + Q1, N);
  finalize_kernel<<<1, 256, 0, stream>>>(wsf + S1, wsf + Q1, bn1_g, bn1_b, b1,
                                         wsf + A1, wsf + C1, 64, invN);
  layer1_apply_kernel<<<2048, 128, 0, stream>>>(pt_fea, W1, wsf + A0, wsf + C0,
                                                wsf + A1, wsf + C1, H1, N);
  // layer2 (WMMA 64->128)
  gemm_stats_kernel<64, 128><<<2048, 128, 0, stream>>>(H1, W2t, b2,
                                                       wsf + S2, wsf + Q2, N);
  finalize_kernel<<<1, 256, 0, stream>>>(wsf + S2, wsf + Q2, bn2_g, bn2_b, b2,
                                         wsf + A2, wsf + C2, 128, invN);
  gemm_apply_kernel<64, 128><<<2048, 128, 0, stream>>>(H1, W2t, wsf + A2, wsf + C2,
                                                       H2, N);
  // layer3 stats (WMMA 128->256)
  gemm_stats_kernel<128, 256><<<2048, 128, 0, stream>>>(H2, W3t, b3,
                                                        wsf + S3, wsf + Q3, N);
  finalize_kernel<<<1, 256, 0, stream>>>(wsf + S3, wsf + Q3, bn3_g, bn3_b, b3,
                                         wsf + A3, wsf + C3, 256, invN);
  // fused layer3-apply + layer4 + segment-max
  final_fused_kernel<<<2048, 128, 0, stream>>>(H2, W3t, wsf + A3, wsf + C3,
                                               W4t, b4, unq_inv, pooled, N);
}
